// PointNet_scan_8761733284514
// MI455X (gfx1250) — matrix operand order repaired and checked by reading, not compiled, not run on hardware
//
#include <hip/hip_runtime.h>
#include <cmath>
#include <cstdint>

// ===========================================================================
// PointNet-scan forward for MI455X (gfx1250, wave32, WMMA).
// GEMM layers -> v_wmma_f32_16x16x32_f16 (f16 in / f32 accum), register-blocked
// 16(O)x64(N) per wave (4 WMMA accumulators sharing one A fragment), fused
// bias+BN+ReLU epilogues, concat-free channel-slab buffers, LDS top-k/kNN.
// ===========================================================================

typedef __attribute__((ext_vector_type(16))) _Float16 v16h;
typedef __attribute__((ext_vector_type(8)))  float    v8f;

#define BSZ  4
#define NPTS 8192
#define MNOD 2048

// ---------------------------------------------------------------------------
// WMMA fused 1x1-conv:  Y[b, cout0+o, n] = act((sum_c W[o,wcol0+c] * X[b,cin0+c,n]
//                                   + lbias[o] + ebias[b,o]) * scale[o] + bias[o])
// One wave computes one 16(O) x 64(N) tile (4 WMMA tiles, shared A fragment);
// 4 waves per block -> each block covers 256 points.
// ---------------------------------------------------------------------------
__global__ __launch_bounds__(128)
void wmma_conv_kernel(const float* __restrict__ X, int cinTot, int cin0,
                      const float* __restrict__ W, int ldw, int wcol0,
                      float* __restrict__ Y, int coutTot, int cout0,
                      const float* __restrict__ lbias,
                      const float* __restrict__ scale,
                      const float* __restrict__ bias,
                      const float* __restrict__ ebias,
                      int O, int Cin, int Npts, int relu)
{
  const int wave = threadIdx.x >> 5;
  const int lane = threadIdx.x & 31;
  const int lh   = lane >> 4;        // lane half (0/1)
  const int lr   = lane & 15;        // row/col within tile
  const int n0 = (blockIdx.x * 4 + wave) * 64;   // 64 points per wave
  const int o0 = blockIdx.y * 16;
  const int b  = blockIdx.z;

  const float* xb   = X + ((size_t)b * cinTot + cin0) * Npts;
  const float* wrow = W + (size_t)(o0 + lr) * ldw + wcol0;

  v8f acc0 = {}, acc1 = {}, acc2 = {}, acc3 = {};
  for (int k = 0; k < Cin; k += 32) {
    // A (16x32 f16): lane lr holds row lr; K split per ISA 7.12.2
    v16h a;
#pragma unroll
    for (int v = 0; v < 8; ++v) {
      const int ka = k + (v & 3) * 2 + (v >> 2) * 16 + lh * 8;
      a[2 * v]     = (_Float16)wrow[ka];
      a[2 * v + 1] = (_Float16)wrow[ka + 1];
    }
    // B (32x16 f16) x4 sub-tiles: lane lr holds column n0+t*16+lr
    v16h bm[4];
#pragma unroll
    for (int t = 0; t < 4; ++t) {
#pragma unroll
      for (int v = 0; v < 8; ++v) {
        const int kb = k + (v & 3) * 2 + (v >> 2) * 16 + lh * 8;
        const float* xp = xb + (size_t)kb * Npts + n0 + t * 16 + lr;
        bm[t][2 * v]     = (_Float16)xp[0];
        bm[t][2 * v + 1] = (_Float16)xp[Npts];
      }
    }
    if (k + 32 < Cin)  // hint next K-slab into cache (global_prefetch_b8)
      __builtin_prefetch(xb + (size_t)(k + 32 + lh * 16) * Npts + n0 + lr, 0, 1);
    acc0 = __builtin_amdgcn_wmma_f32_16x16x32_f16(false, a, false, bm[0],
                                                  (short)0, acc0, false, false);
    acc1 = __builtin_amdgcn_wmma_f32_16x16x32_f16(false, a, false, bm[1],
                                                  (short)0, acc1, false, false);
    acc2 = __builtin_amdgcn_wmma_f32_16x16x32_f16(false, a, false, bm[2],
                                                  (short)0, acc2, false, false);
    acc3 = __builtin_amdgcn_wmma_f32_16x16x32_f16(false, a, false, bm[3],
                                                  (short)0, acc3, false, false);
  }

  float* yb = Y + ((size_t)b * coutTot + cout0) * Npts;
#pragma unroll
  for (int v = 0; v < 8; ++v) {
    const int oc = o0 + lh * 8 + v;          // C/D: VGPR v -> row v + 8*half
    float lb0 = lbias ? lbias[oc] : 0.f;
    if (ebias) lb0 += ebias[b * O + oc];
    const float sc = scale ? scale[oc] : 1.f;
    const float bi = scale ? bias[oc]  : 0.f;
    float* yr = yb + (size_t)oc * Npts + n0 + lr;
    float r0 = (acc0[v] + lb0) * sc + bi;
    float r1 = (acc1[v] + lb0) * sc + bi;
    float r2 = (acc2[v] + lb0) * sc + bi;
    float r3 = (acc3[v] + lb0) * sc + bi;
    if (relu) { r0 = fmaxf(r0, 0.f); r1 = fmaxf(r1, 0.f);
                r2 = fmaxf(r2, 0.f); r3 = fmaxf(r3, 0.f); }
    yr[0]  = r0;
    yr[16] = r1;
    yr[32] = r2;
    yr[48] = r3;
  }
}

// ---------------------------------------------------------------------------
// Simple conv for tiny Cin/O (conv1: 3->64, conv10: 128->2)
// ---------------------------------------------------------------------------
__global__ void conv_simple_kernel(const float* __restrict__ X, int cinTot, int cin0,
                                   const float* __restrict__ W, int ldw,
                                   float* __restrict__ Y, int coutTot, int cout0,
                                   const float* __restrict__ scale,
                                   const float* __restrict__ bias,
                                   int O, int Cin, int Npts, int relu)
{
  size_t i = (size_t)blockIdx.x * blockDim.x + threadIdx.x;
  size_t total = (size_t)BSZ * O * Npts;
  if (i >= total) return;
  int n = (int)(i % Npts);
  int o = (int)((i / Npts) % O);
  int b = (int)(i / ((size_t)Npts * O));
  const float* xb = X + ((size_t)b * cinTot + cin0) * Npts + n;
  const float* wr = W + (size_t)o * ldw;
  float acc = 0.f;
  for (int c = 0; c < Cin; ++c) acc += wr[c] * xb[(size_t)c * Npts];
  if (scale) acc = acc * scale[o] + bias[o];
  if (relu)  acc = fmaxf(acc, 0.f);
  Y[((size_t)b * coutTot + cout0 + o) * Npts + n] = acc;
}

// ---------------------------------------------------------------------------
// Row max:  rows of length L contiguous; row = b*C + c -> out[b*oStride+oOff+c]
// ---------------------------------------------------------------------------
__global__ __launch_bounds__(256)
void rowmax_kernel(const float* __restrict__ X, float* __restrict__ out,
                   int L, int C, int oStride, int oOff)
{
  __shared__ float red[256];
  const float* row = X + (size_t)blockIdx.x * L;
  float m = -INFINITY;
  for (int i = threadIdx.x; i < L; i += 256) m = fmaxf(m, row[i]);
  red[threadIdx.x] = m;
  __syncthreads();
  for (int s = 128; s > 0; s >>= 1) {
    if (threadIdx.x < s) red[threadIdx.x] = fmaxf(red[threadIdx.x], red[threadIdx.x + s]);
    __syncthreads();
  }
  if (threadIdx.x == 0) {
    int c = blockIdx.x % C, b = blockIdx.x / C;
    out[(size_t)b * oStride + oOff + c] = red[0];
  }
}

// ---------------------------------------------------------------------------
// scores[b,e,n] = sigmoid(sum_c feat[b,c,n] * vec[b, c*8+e])
// ---------------------------------------------------------------------------
__global__ void scores_kernel(const float* __restrict__ feat,
                              const float* __restrict__ vec,
                              float* __restrict__ scores, int Npts)
{
  int n = blockIdx.x * blockDim.x + threadIdx.x;
  int b = blockIdx.y;
  if (n >= Npts) return;
  float s[8] = {0.f, 0.f, 0.f, 0.f, 0.f, 0.f, 0.f, 0.f};
  const float* fb = feat + (size_t)b * 128 * Npts + n;
  const float* vb = vec + (size_t)b * 1024;
  for (int c = 0; c < 128; ++c) {
    float f = fb[(size_t)c * Npts];
#pragma unroll
    for (int e = 0; e < 8; ++e) s[e] += f * vb[c * 8 + e];
  }
#pragma unroll
  for (int e = 0; e < 8; ++e)
    scores[((size_t)b * 8 + e) * Npts + n] = 1.f / (1.f + expf(-s[e]));
}

// ---------------------------------------------------------------------------
// Per (b,e): top-256 of scores row (desc, tie -> lowest index).
// out ordering matches torch/jax reshape: j = r*8 + e
// ---------------------------------------------------------------------------
__global__ __launch_bounds__(256)
void topk256_kernel(const float* __restrict__ scores, int Npts,
                    int* __restrict__ idxOut, float* __restrict__ valOut)
{
  __shared__ float vals[NPTS];
  __shared__ float rv[256];
  __shared__ int   ri[256];
  const float* src = scores + (size_t)blockIdx.x * Npts;
  for (int i = threadIdx.x; i < Npts; i += 256) vals[i] = src[i];
  __syncthreads();
  int b = blockIdx.x >> 3, e = blockIdx.x & 7;
  for (int r = 0; r < 256; ++r) {
    float bv = -INFINITY; int bi = 0x7fffffff;
    for (int i = threadIdx.x; i < Npts; i += 256) {
      float v = vals[i];
      if (v > bv || (v == bv && i < bi)) { bv = v; bi = i; }
    }
    rv[threadIdx.x] = bv; ri[threadIdx.x] = bi;
    __syncthreads();
    for (int s = 128; s > 0; s >>= 1) {
      if (threadIdx.x < s) {
        float ov = rv[threadIdx.x + s]; int oi = ri[threadIdx.x + s];
        if (ov > rv[threadIdx.x] || (ov == rv[threadIdx.x] && oi < ri[threadIdx.x])) {
          rv[threadIdx.x] = ov; ri[threadIdx.x] = oi;
        }
      }
      __syncthreads();
    }
    if (threadIdx.x == 0) {
      int w = ri[0];
      idxOut[(size_t)b * MNOD + r * 8 + e] = w;
      valOut[(size_t)b * MNOD + r * 8 + e] = rv[0];
      vals[w] = -INFINITY;
    }
    __syncthreads();
  }
}

// ---------------------------------------------------------------------------
// node/node_static gather (also writes the two graph outputs in d_out)
// ---------------------------------------------------------------------------
__global__ void gather_nodes_kernel(const float* __restrict__ xyz,
                                    const int* __restrict__ idx,
                                    const float* __restrict__ val,
                                    float* __restrict__ node,
                                    float* __restrict__ nstat,
                                    float* __restrict__ out_node,
                                    float* __restrict__ out_nstat,
                                    int Npts, int M)
{
  int j = blockIdx.x * blockDim.x + threadIdx.x;
  int b = blockIdx.y;
  if (j >= M) return;
  int id = idx[(size_t)b * M + j];
  float v = val[(size_t)b * M + j];
#pragma unroll
  for (int d = 0; d < 3; ++d) {
    float g = xyz[((size_t)b * 3 + d) * Npts + id];
    size_t o = ((size_t)b * 3 + d) * M + j;
    nstat[o] = g;       node[o] = g * v;
    out_nstat[o] = g;   out_node[o] = g * v;
  }
}

// node_feature[b,c,j] = feat[b,c,idx]*val  -> written into a channel slab
__global__ void gather_feat_kernel(const float* __restrict__ feat, int cfTot, int cf0, int C,
                                   const int* __restrict__ idx, const float* __restrict__ val,
                                   float* __restrict__ out, int coTot, int co0,
                                   int Npts, int M)
{
  size_t i = (size_t)blockIdx.x * blockDim.x + threadIdx.x;
  size_t total = (size_t)BSZ * C * M;
  if (i >= total) return;
  int j = (int)(i % M);
  int c = (int)((i / M) % C);
  int b = (int)(i / ((size_t)M * C));
  int id = idx[(size_t)b * M + j];
  out[((size_t)b * coTot + co0 + c) * M + j] =
      feat[((size_t)b * cfTot + cf0 + c) * Npts + id] * val[(size_t)b * M + j];
}

// ---------------------------------------------------------------------------
// aggregate: per (b,m) find 20 nearest input points, max-pool feat over them
// ---------------------------------------------------------------------------
__global__ __launch_bounds__(256)
void aggregate_kernel(const float* __restrict__ xyz, const float* __restrict__ node,
                      const float* __restrict__ feat,
                      float* __restrict__ out, int coTot, int co0,
                      int Npts, int M, int knn)
{
  __shared__ float d[NPTS];
  __shared__ float rv[256];
  __shared__ int   ri[256];
  __shared__ int   nbr[32];
  int m = blockIdx.x, b = blockIdx.y;
  float nx = node[((size_t)b * 3 + 0) * M + m];
  float ny = node[((size_t)b * 3 + 1) * M + m];
  float nz = node[((size_t)b * 3 + 2) * M + m];
  float nn = nx * nx + ny * ny + nz * nz;
  for (int i = threadIdx.x; i < Npts; i += 256) {
    float px = xyz[((size_t)b * 3 + 0) * Npts + i];
    float py = xyz[((size_t)b * 3 + 1) * Npts + i];
    float pz = xyz[((size_t)b * 3 + 2) * Npts + i];
    d[i] = px * px + py * py + pz * pz + nn - 2.f * (px * nx + py * ny + pz * nz);
  }
  __syncthreads();
  for (int r = 0; r < knn; ++r) {
    float bv = INFINITY; int bi = 0x7fffffff;
    for (int i = threadIdx.x; i < Npts; i += 256) {
      float v = d[i];
      if (v < bv || (v == bv && i < bi)) { bv = v; bi = i; }
    }
    rv[threadIdx.x] = bv; ri[threadIdx.x] = bi;
    __syncthreads();
    for (int s = 128; s > 0; s >>= 1) {
      if (threadIdx.x < s) {
        float ov = rv[threadIdx.x + s]; int oi = ri[threadIdx.x + s];
        if (ov < rv[threadIdx.x] || (ov == rv[threadIdx.x] && oi < ri[threadIdx.x])) {
          rv[threadIdx.x] = ov; ri[threadIdx.x] = oi;
        }
      }
      __syncthreads();
    }
    if (threadIdx.x == 0) { nbr[r] = ri[0]; d[ri[0]] = INFINITY; }
    __syncthreads();
  }
  for (int c = threadIdx.x; c < 128; c += 256) {
    float mx = -INFINITY;
    for (int k = 0; k < knn; ++k)
      mx = fmaxf(mx, feat[((size_t)b * 128 + c) * Npts + nbr[k]]);
    out[((size_t)b * coTot + co0 + c) * M + m] = mx;
  }
}

// ---------------------------------------------------------------------------
// unpool: 3-NN nodes per point; w = softmax over selected sq-dists (faithful)
// ---------------------------------------------------------------------------
__global__ __launch_bounds__(256)
void unpool_kernel(const float* __restrict__ xyz, const float* __restrict__ node,
                   const float* __restrict__ feat,       // [B,128,M]
                   float* __restrict__ out, int coTot,   // write ch 0..127
                   int Npts, int M)
{
  __shared__ float d[MNOD];
  __shared__ float rv[256];
  __shared__ int   ri[256];
  __shared__ int   sid[3];
  __shared__ float sdv[3];
  int n = blockIdx.x, b = blockIdx.y;
  float px = xyz[((size_t)b * 3 + 0) * Npts + n];
  float py = xyz[((size_t)b * 3 + 1) * Npts + n];
  float pz = xyz[((size_t)b * 3 + 2) * Npts + n];
  float pp = px * px + py * py + pz * pz;
  for (int m = threadIdx.x; m < M; m += 256) {
    float qx = node[((size_t)b * 3 + 0) * M + m];
    float qy = node[((size_t)b * 3 + 1) * M + m];
    float qz = node[((size_t)b * 3 + 2) * M + m];
    d[m] = pp + qx * qx + qy * qy + qz * qz - 2.f * (px * qx + py * qy + pz * qz);
  }
  __syncthreads();
  for (int r = 0; r < 3; ++r) {
    float bv = INFINITY; int bi = 0x7fffffff;
    for (int m = threadIdx.x; m < M; m += 256) {
      float v = d[m];
      if (v < bv || (v == bv && m < bi)) { bv = v; bi = m; }
    }
    rv[threadIdx.x] = bv; ri[threadIdx.x] = bi;
    __syncthreads();
    for (int s = 128; s > 0; s >>= 1) {
      if (threadIdx.x < s) {
        float ov = rv[threadIdx.x + s]; int oi = ri[threadIdx.x + s];
        if (ov < rv[threadIdx.x] || (ov == rv[threadIdx.x] && oi < ri[threadIdx.x])) {
          rv[threadIdx.x] = ov; ri[threadIdx.x] = oi;
        }
      }
      __syncthreads();
    }
    if (threadIdx.x == 0) { sid[r] = ri[0]; sdv[r] = rv[0]; d[ri[0]] = INFINITY; }
    __syncthreads();
  }
  // softmax over the raw selected sq-distances (reference-faithful)
  float mx = fmaxf(sdv[0], fmaxf(sdv[1], sdv[2]));
  float w0 = expf(sdv[0] - mx), w1 = expf(sdv[1] - mx), w2 = expf(sdv[2] - mx);
  float inv = 1.f / (w0 + w1 + w2);
  w0 *= inv; w1 *= inv; w2 *= inv;
  int i0 = sid[0], i1 = sid[1], i2 = sid[2];
  for (int c = threadIdx.x; c < 128; c += 256) {
    const float* fc = feat + ((size_t)b * 128 + c) * M;
    out[((size_t)b * coTot + c) * Npts + n] = w0 * fc[i0] + w1 * fc[i1] + w2 * fc[i2];
  }
}

// ---------------------------------------------------------------------------
// tiny GEMV / fused linear head: out[b,o]=act((in.w + lb)*s + bb)
// ---------------------------------------------------------------------------
__global__ void linear_kernel(const float* __restrict__ in, const float* __restrict__ W,
                              int ldw, int wcol0,
                              const float* __restrict__ lb,
                              const float* __restrict__ s, const float* __restrict__ bb,
                              float* __restrict__ out, int O, int L, int relu)
{
  int i = blockIdx.x * blockDim.x + threadIdx.x;
  if (i >= BSZ * O) return;
  int o = i % O, b = i / O;
  const float* w = W + (size_t)o * ldw + wcol0;
  const float* x = in + (size_t)b * L;
  float acc = 0.f;
  for (int l = 0; l < L; ++l) acc += w[l] * x[l];
  if (lb) acc += lb[o];
  if (s)  acc = acc * s[o] + bb[o];
  if (relu) acc = fmaxf(acc, 0.f);
  out[(size_t)b * O + o] = acc;
}

// channel-slab copy (builds concats)
__global__ void copy_ch_kernel(const float* __restrict__ src, int csTot, int cs0,
                               float* __restrict__ dst, int cdTot, int cd0,
                               int C, int Npts)
{
  size_t i = (size_t)blockIdx.x * blockDim.x + threadIdx.x;
  size_t total = (size_t)BSZ * C * Npts;
  if (i >= total) return;
  int n = (int)(i % Npts);
  int c = (int)((i / Npts) % C);
  int b = (int)(i / ((size_t)Npts * C));
  dst[((size_t)b * cdTot + cd0 + c) * Npts + n] =
      src[((size_t)b * csTot + cs0 + c) * Npts + n];
}

// ===========================================================================
// Host side
// ===========================================================================
static void wmma_conv(const float* X, int cinTot, int cin0, const float* W, int ldw,
                      int wcol0, float* Y, int coutTot, int cout0, const float* lbias,
                      const float* scale, const float* bias, const float* ebias,
                      int O, int Cin, int Npts, int relu, hipStream_t st)
{
  dim3 g(Npts / 256, O / 16, BSZ);   // 4 waves x 64 points each per block
  wmma_conv_kernel<<<g, 128, 0, st>>>(X, cinTot, cin0, W, ldw, wcol0, Y, coutTot,
                                      cout0, lbias, scale, bias, ebias, O, Cin, Npts, relu);
}

extern "C" void kernel_launch(void* const* d_in, const int* in_sizes, int n_in,
                              void* d_out, int out_size, void* d_ws, size_t ws_size,
                              hipStream_t stream)
{
  (void)in_sizes; (void)n_in; (void)out_size;
  const float* x        = (const float*)d_in[0];
  const float* conv1_w  = (const float*)d_in[1];
  const float* conv2_w  = (const float*)d_in[2];
  const float* conv2m_w = (const float*)d_in[3];
  const float* conv3_w  = (const float*)d_in[4];
  const float* conv4_w  = (const float*)d_in[5];
  const float* conv5_w  = (const float*)d_in[6];
  const float* conv6_w  = (const float*)d_in[7];
  const float* conv7_w  = (const float*)d_in[8];
  const float* conv8_w  = (const float*)d_in[9];
  const float* conv9_w  = (const float*)d_in[10];
  const float* conv10_w = (const float*)d_in[11];
  const float* bn1_s  = (const float*)d_in[12], * bn1_b  = (const float*)d_in[13];
  const float* bn2_s  = (const float*)d_in[14], * bn2_b  = (const float*)d_in[15];
  const float* bn2m_s = (const float*)d_in[16], * bn2m_b = (const float*)d_in[17];
  const float* bn3_s  = (const float*)d_in[18], * bn3_b  = (const float*)d_in[19];
  const float* bn4_s  = (const float*)d_in[20], * bn4_b  = (const float*)d_in[21];
  const float* bn5_s  = (const float*)d_in[22], * bn5_b  = (const float*)d_in[23];
  const float* bn6c_s = (const float*)d_in[24], * bn6c_b = (const float*)d_in[25];
  const float* bn7c_s = (const float*)d_in[26], * bn7c_b = (const float*)d_in[27];
  const float* bn8_s  = (const float*)d_in[28], * bn8_b  = (const float*)d_in[29];
  const float* bn9_s  = (const float*)d_in[30], * bn9_b  = (const float*)d_in[31];
  const float* bn6h_s = (const float*)d_in[32], * bn6h_b = (const float*)d_in[33];
  const float* bn7h_s = (const float*)d_in[34], * bn7h_b = (const float*)d_in[35];
  const float* pool_w = (const float*)d_in[36], * pool_b = (const float*)d_in[37];
  const float* lin1_w = (const float*)d_in[38];
  const float* lin2_w = (const float*)d_in[39], * lin2_b = (const float*)d_in[40];
  const float* lin3_w = (const float*)d_in[41], * lin3_b = (const float*)d_in[42];

  // ---- workspace layout (floats) ----
  const size_t O_X12   = 0;                              // [B,128,N]
  const size_t O_BIGA  = O_X12 + (size_t)BSZ*128*NPTS;   // [B,1024,N] arena
  const size_t O_SCOR  = O_BIGA + (size_t)BSZ*1024*NPTS; // [B,8,N]
  const size_t O_VEC   = O_SCOR + (size_t)BSZ*8*NPTS;    // [B,1024]
  const size_t O_VECT  = O_VEC  + (size_t)BSZ*1024;      // [B,2048]
  const size_t O_BIAS6 = O_VECT + (size_t)BSZ*2048;      // [B,128]
  const size_t O_C1    = O_BIAS6 + (size_t)BSZ*128;      // [B,512]
  const size_t O_C2    = O_C1 + (size_t)BSZ*512;         // [B,256]
  const size_t O_TIDX  = O_C2 + (size_t)BSZ*256;         // [B,2048] int
  const size_t O_TVAL  = O_TIDX + (size_t)BSZ*MNOD;      // [B,2048]
  const size_t O_NODE  = O_TVAL + (size_t)BSZ*MNOD;      // [B,3,M]
  const size_t O_NSTAT = O_NODE + (size_t)BSZ*3*MNOD;    // [B,3,M]
  const size_t TOTAL   = O_NSTAT + (size_t)BSZ*3*MNOD;
  if (ws_size < TOTAL * sizeof(float)) return;
  float* ws = (float*)d_ws;
  int*   wsi = (int*)d_ws;

  // arena sub-buffers (non-overlapping lifetimes)
  float* X12  = ws + O_X12;
  float* BIGA = ws + O_BIGA;                      // xt1 -> proj -> slabs below
  float* H    = BIGA + 0;                         // [B,256,M]
  float* X34  = BIGA + (size_t)BSZ*256*MNOD;      // [B,512,M]
  float* XT2  = X34  + (size_t)BSZ*512*MNOD;      // [B,1024,M]
  float* CAT7 = XT2  + (size_t)BSZ*1024*MNOD;     // [B,384,M]
  float* S7   = CAT7 + (size_t)BSZ*384*MNOD;      // [B,128,M]
  float* CAT8 = S7   + (size_t)BSZ*128*MNOD;      // [B,192,N]
  float* CAT9 = CAT8 + (size_t)BSZ*192*NPTS;      // [B,192,N]
  float* S9   = XT2;                              // [B,128,N] (xt2 dead by then)
  float* SCOR = ws + O_SCOR;
  float* VEC  = ws + O_VEC;
  float* VECT = ws + O_VECT;
  float* BIAS6= ws + O_BIAS6;
  float* C1   = ws + O_C1;
  float* C2   = ws + O_C2;
  int*   TIDX = wsi + O_TIDX;
  float* TVAL = ws + O_TVAL;
  float* NODE = ws + O_NODE;
  float* NSTAT= ws + O_NSTAT;

  float* out        = (float*)d_out;
  float* out_cls    = out;                 // [B,15]
  float* out_seg    = out + 60;            // [B,2,N]
  float* out_node   = out + 60 + (size_t)BSZ*2*NPTS;            // [B,3,M]
  float* out_nstat  = out_node + (size_t)BSZ*3*MNOD;            // [B,3,M]

  // 1) x1 = relu(bn1(conv1 x))  -> X12 ch 0..63
  {
    size_t tot = (size_t)BSZ*64*NPTS;
    conv_simple_kernel<<<(tot+255)/256, 256, 0, stream>>>(
        x, 3, 0, conv1_w, 3, X12, 128, 0, bn1_s, bn1_b, 64, 3, NPTS, 1);
  }
  // 2) x2 = relu(bn2(conv2 x1)) -> X12 ch 64..127
  wmma_conv(X12, 128, 0, conv2_w, 64, 0, X12, 128, 64,
            nullptr, bn2_s, bn2_b, nullptr, 64, 64, NPTS, 1, stream);
  // 3) xt1 = relu(bn2m(conv2m xt1_)) -> BIGA [B,1024,N]
  wmma_conv(X12, 128, 0, conv2m_w, 128, 0, BIGA, 1024, 0,
            nullptr, bn2m_s, bn2m_b, nullptr, 1024, 128, NPTS, 1, stream);
  // 4) vector[:,0:1024] = xt1.max(-1)
  rowmax_kernel<<<BSZ*1024, 256, 0, stream>>>(BIGA, VECT, NPTS, 1024, 2048, 0);
  // 5) pool proj = conv(pool_w, xt1_) + pool_b -> BIGA (reused)
  wmma_conv(X12, 128, 0, pool_w, 128, 0, BIGA, 1024, 0,
            pool_b, nullptr, nullptr, nullptr, 1024, 128, NPTS, 0, stream);
  // 6) vec = proj.max(-1)  [B,1024]
  rowmax_kernel<<<BSZ*1024, 256, 0, stream>>>(BIGA, VEC, NPTS, 1024, 1024, 0);
  // 7) scores [B,8,N]
  scores_kernel<<<dim3((NPTS+255)/256, BSZ), 256, 0, stream>>>(X12, VEC, SCOR, NPTS);
  // 8) per-(b,e) top-256
  topk256_kernel<<<BSZ*8, 256, 0, stream>>>(SCOR, NPTS, TIDX, TVAL);
  // 9) node / node_static gathers (+ graph outputs)
  gather_nodes_kernel<<<dim3((MNOD+255)/256, BSZ), 256, 0, stream>>>(
      x, TIDX, TVAL, NODE, NSTAT, out_node, out_nstat, NPTS, MNOD);
  // 10) nf1 -> H ch 0..127
  {
    size_t tot = (size_t)BSZ*128*MNOD;
    gather_feat_kernel<<<(tot+255)/256, 256, 0, stream>>>(
        X12, 128, 0, 128, TIDX, TVAL, H, 256, 0, NPTS, MNOD);
  }
  // 11) aggregate -> H ch 128..255
  aggregate_kernel<<<dim3(MNOD, BSZ), 256, 0, stream>>>(
      x, NODE, X12, H, 256, 128, NPTS, MNOD, 20);
  // 12) x3 = relu(bn3(conv3 h)) -> X34 ch 0..255
  wmma_conv(H, 256, 0, conv3_w, 256, 0, X34, 512, 0,
            nullptr, bn3_s, bn3_b, nullptr, 256, 256, MNOD, 1, stream);
  // 13) x4 -> X34 ch 256..511
  wmma_conv(X34, 512, 0, conv4_w, 256, 0, X34, 512, 256,
            nullptr, bn4_s, bn4_b, nullptr, 256, 256, MNOD, 1, stream);
  // 14) xt2 = relu(bn5(conv5 [x3;x4]))
  wmma_conv(X34, 512, 0, conv5_w, 512, 0, XT2, 1024, 0,
            nullptr, bn5_s, bn5_b, nullptr, 1024, 512, MNOD, 1, stream);
  // 15) vector[:,1024:2048] = xt2.max(-1)
  rowmax_kernel<<<BSZ*1024, 256, 0, stream>>>(XT2, VECT, MNOD, 1024, 2048, 1024);
  // 16) classification head
  linear_kernel<<<(BSZ*512+63)/64, 64, 0, stream>>>(
      VECT, lin1_w, 2048, 0, nullptr, bn6h_s, bn6h_b, C1, 512, 2048, 1);
  linear_kernel<<<(BSZ*256+63)/64, 64, 0, stream>>>(
      C1, lin2_w, 512, 0, lin2_b, bn7h_s, bn7h_b, C2, 256, 512, 1);
  linear_kernel<<<(BSZ*15+63)/64, 64, 0, stream>>>(
      C2, lin3_w, 256, 0, lin3_b, nullptr, nullptr, out_cls, 15, 256, 0);
  // 17) fold vrep into per-(b,o) bias: bias6 = W6[:, :2048] @ vector
  linear_kernel<<<(BSZ*128+63)/64, 64, 0, stream>>>(
      VECT, conv6_w, 2304, 0, nullptr, nullptr, nullptr, BIAS6, 128, 2048, 0);
  // 18) s = relu(bn6c(conv6 [vrep;x4])) -> CAT7 ch 0..127
  wmma_conv(X34, 512, 256, conv6_w, 2304, 2048, CAT7, 384, 0,
            nullptr, bn6c_s, bn6c_b, BIAS6, 128, 256, MNOD, 1, stream);
  // 19) x3 -> CAT7 ch 128..383
  {
    size_t tot = (size_t)BSZ*256*MNOD;
    copy_ch_kernel<<<(tot+255)/256, 256, 0, stream>>>(X34, 512, 0, CAT7, 384, 128, 256, MNOD);
  }
  // 20) s = relu(bn7c(conv7 cat7)) -> S7
  wmma_conv(CAT7, 384, 0, conv7_w, 384, 0, S7, 128, 0,
            nullptr, bn7c_s, bn7c_b, nullptr, 128, 384, MNOD, 1, stream);
  // 21) unpool -> CAT8 ch 0..127
  unpool_kernel<<<dim3(NPTS, BSZ), 256, 0, stream>>>(x, NODE, S7, CAT8, 192, NPTS, MNOD);
  // 22) x2 -> CAT8 ch 128..191
  {
    size_t tot = (size_t)BSZ*64*NPTS;
    copy_ch_kernel<<<(tot+255)/256, 256, 0, stream>>>(X12, 128, 64, CAT8, 192, 128, 64, NPTS);
  }
  // 23) s = relu(bn8(conv8 cat8)) -> CAT9 ch 0..127
  wmma_conv(CAT8, 192, 0, conv8_w, 192, 0, CAT9, 192, 0,
            nullptr, bn8_s, bn8_b, nullptr, 128, 192, NPTS, 1, stream);
  // 24) x1 -> CAT9 ch 128..191
  {
    size_t tot = (size_t)BSZ*64*NPTS;
    copy_ch_kernel<<<(tot+255)/256, 256, 0, stream>>>(X12, 128, 0, CAT9, 192, 128, 64, NPTS);
  }
  // 25) s = relu(bn9(conv9 cat9)) -> S9
  wmma_conv(CAT9, 192, 0, conv9_w, 192, 0, S9, 128, 0,
            nullptr, bn9_s, bn9_b, nullptr, 128, 192, NPTS, 1, stream);
  // 26) logits_seg = conv10(s) -> d_out
  {
    size_t tot = (size_t)BSZ*2*NPTS;
    conv_simple_kernel<<<(tot+255)/256, 256, 0, stream>>>(
        S9, 128, 0, conv10_w, 128, out_seg, 2, 0, nullptr, nullptr, 2, 128, NPTS, 0);
  }
}
